// BitwiseOperator_77773267796517
// MI455X (gfx1250) — matrix operand order repaired and checked by reading, not compiled, not run on hardware
//
#include <hip/hip_runtime.h>

typedef _Float16 h2   __attribute__((ext_vector_type(2)));
typedef _Float16 v16h __attribute__((ext_vector_type(16)));
typedef float    v8f  __attribute__((ext_vector_type(8)));
typedef float    v4f  __attribute__((ext_vector_type(4)));

union AFrag {
  v16h v;
  h2   p[8];
  unsigned int u[8];
};

// One wave (32 lanes) owns a tile of 16 batch rows for the full 64-digit scan.
// GEMM: D[b,v] = sum_k X[b,k] * S[k,v] via v_wmma_f32_16x16x32_f16, K=256 in 8 steps.
// K-step s covers i in {2s,2s+1}; j is cyclically remapped by -/+2s so the one-hot
// B fragments (u%16 / d%16 selectors) are identical for every k-step.
__global__ __launch_bounds__(256, 1) void probnum_addsub_kernel(
    const float* __restrict__ op1,
    const float* __restrict__ op2,
    float* __restrict__ outA,
    float* __restrict__ outS)
{
  const int lane = threadIdx.x & 31;
  const int wv   = threadIdx.x >> 5;
  const int tile = blockIdx.x * 8 + wv;     // 2048 tiles total
  const int row  = lane & 15;               // batch row for loads / digit v for stores
  const int hi   = lane >> 4;               // K-half (B) / i offset, i = 2s+hi (A)

  // ---- static one-hot B fragments (k-step invariant) ----
  // B element e (K = e + 16*hi): i = 2s+hi, j_add=(e-2s)%16 -> u%16 = (hi+e)%16
  //                              j_sub=(e+2s)%16 -> d%16 = (hi-e)%16
  AFrag B0, B1;
  {
    const int e0 = (row - hi) & 15;   // add-digit column selector
    const int e1 = (hi - row) & 15;   // sub-digit column selector
#pragma unroll
    for (int g = 0; g < 8; ++g) {
      B0.u[g] = ((e0 >> 1) == g) ? ((e0 & 1) ? 0x3C000000u : 0x00003C00u) : 0u;
      B1.u[g] = ((e1 >> 1) == g) ? ((e1 & 1) ? 0x3C000000u : 0x00003C00u) : 0u;
    }
  }

  // lane-rotation indices (within each 16-lane half), byte units for ds_bpermute
  const int idx_m1 = (((lane & 16) | ((lane - 1) & 15)) << 2);
  const int idx_p1 = (((lane & 16) | ((lane + 1) & 15)) << 2);
  int idx_row[8];
#pragma unroll
  for (int r = 0; r < 8; ++r) idx_row[r] = ((hi << 3) + r) << 2;  // scalar->D-layout bcast

  const int brow = tile * 16 + row;
  const float* pa = op1 + (size_t)brow * 1024;   // 64 digits * 16 values
  const float* pb = op2 + (size_t)brow * 1024;
  float* poa = outA + (size_t)(tile * 16 + hi * 8) * 1024 + row;
  float* pos = outS + (size_t)(tile * 16 + hi * 8) * 1024 + row;

  // preload digit 0 rows
  v4f ra[4], rb[4];
#pragma unroll
  for (int c = 0; c < 4; ++c) {
    ra[c] = *(const v4f*)(pa + c * 4);
    rb[c] = *(const v4f*)(pb + c * 4);
  }

  float c1s = 0.0f, b1s = 0.0f;   // P(carry=1), P(borrow=1); init dist = [1,0]

#pragma unroll 1
  for (int l = 0; l < 64; ++l) {
    // ---- unpack current rows ----
    float a[16], bb[16];
#pragma unroll
    for (int c = 0; c < 4; ++c) {
#pragma unroll
      for (int k = 0; k < 4; ++k) { a[c*4+k] = ra[c][k]; bb[c*4+k] = rb[c][k]; }
    }

    // ---- f16 operands for the A (data) side ----
    h2 arep[16];
#pragma unroll
    for (int i = 0; i < 16; ++i) { _Float16 h = (_Float16)a[i]; h2 t; t.x = h; t.y = h; arep[i] = t; }
    h2 bq[8];
#pragma unroll
    for (int q = 0; q < 8; ++q) { h2 t; t.x = (_Float16)bb[2*q]; t.y = (_Float16)bb[2*q+1]; bq[q] = t; }


    // ---- per-lane scalar recurrence quantities (f32, exact) ----
    float Sb[16];
    Sb[15] = bb[15];
#pragma unroll
    for (int k = 14; k >= 0; --k) Sb[k] = bb[k] + Sb[k + 1];
    float hi0 = 0.f, loS = 0.f, c15a = 0.f, c15s = 0.f;
#pragma unroll
    for (int i = 1; i < 16; ++i) hi0 = fmaf(a[i], Sb[16 - i], hi0);   // P(i+j>=16)
#pragma unroll
    for (int i = 0; i < 15; ++i) loS = fmaf(a[i], Sb[i + 1], loS);    // P(i<j)
#pragma unroll
    for (int i = 0; i < 16; ++i) c15a = fmaf(a[i], bb[15 - i], c15a); // P(i+j==15)
#pragma unroll
    for (int i = 0; i < 16; ++i) c15s = fmaf(a[i], bb[i], c15s);      // P(i==j)

    // ---- double-buffer: issue next digit's loads now ----
    const int ln = (l < 63) ? (l + 1) : 63;
#pragma unroll
    for (int c = 0; c < 4; ++c) {
      ra[c] = *(const v4f*)(pa + ln * 16 + c * 4);
      rb[c] = *(const v4f*)(pb + ln * 16 + c * 4);
    }
    const int lp = (l + 8 < 64) ? (l + 8) : 63;
    __builtin_prefetch(pa + lp * 16, 0, 3);
    __builtin_prefetch(pb + lp * 16, 0, 3);

    // ---- WMMA: usum (mod-16 conv) and dsum (mod-16 corr) ----
    v8f accA = {0.f,0.f,0.f,0.f,0.f,0.f,0.f,0.f};
    v8f accS = {0.f,0.f,0.f,0.f,0.f,0.f,0.f,0.f};
#pragma unroll
    for (int s = 0; s < 8; ++s) {
      AFrag fa, fs;
#pragma unroll
      for (int t = 0; t < 8; ++t) {
        const h2 ai = hi ? arep[2 * s + 1] : arep[2 * s];
        fa.p[t] = ai * bq[(t - s) & 7];   // j_add = (jraw - 2s) % 16
        fs.p[t] = ai * bq[(t + s) & 7];   // j_sub = (jraw + 2s) % 16
      }
      accA = __builtin_amdgcn_wmma_f32_16x16x32_f16(false, fa.v, false, B0.v,
                                                    (short)0, accA, false, false);
      accS = __builtin_amdgcn_wmma_f32_16x16x32_f16(false, fs.v, false, B1.v,
                                                    (short)0, accS, false, false);
    }

    // ---- carry/borrow mixing + store ----
    const int c1i = __float_as_int(c1s);
    const int b1i = __float_as_int(b1s);
#pragma unroll
    for (int r = 0; r < 8; ++r) {
      const float us = accA[r];
      const float ds = accS[r];
      const float ru  = __int_as_float(__builtin_amdgcn_ds_bpermute(idx_m1, __float_as_int(us)));
      const float rd  = __int_as_float(__builtin_amdgcn_ds_bpermute(idx_p1, __float_as_int(ds)));
      const float c1b = __int_as_float(__builtin_amdgcn_ds_bpermute(idx_row[r], c1i));
      const float b1b = __int_as_float(__builtin_amdgcn_ds_bpermute(idx_row[r], b1i));
      poa[(size_t)r * 1024 + l * 16] = us + c1b * (ru - us);  // c0*usum + c1*rot(usum)
      pos[(size_t)r * 1024 + l * 16] = ds + b1b * (rd - ds);  // b0*dsum + b1*rot(dsum)
    }

    // ---- recurrence (uses pre-update c1s/b1s above) ----
    c1s = hi0 + c1s * c15a;
    b1s = loS + b1s * c15s;
  }
}

extern "C" void kernel_launch(void* const* d_in, const int* in_sizes, int n_in,
                              void* d_out, int out_size, void* d_ws, size_t ws_size,
                              hipStream_t stream) {
  const float* op1 = (const float*)d_in[0];
  const float* op2 = (const float*)d_in[1];
  // lookup tables d_in[2..5] are one-hot; their structure is folded into the kernel.
  float* out  = (float*)d_out;
  float* outA = out;                                   // add_result [32768,64,16]
  float* outS = out + (size_t)32768 * 64 * 16;         // sub_result [32768,64,16]
  probnum_addsub_kernel<<<dim3(256), dim3(256), 0, stream>>>(op1, op2, outA, outS);
}
